// CoAttention_2138893713842
// MI455X (gfx1250) — matrix operand order repaired and checked by reading, compiled, not run on hardware
//
#include <hip/hip_runtime.h>
#include <hip/hip_bf16.h>
#include <math.h>

// ---------------------------------------------------------------------------
// CDNA5 (gfx1250) co-attention pipeline.
// Big GEMMs via v_wmma_f32_16x16x32_bf16 (wave32, 16x16 tiles, f32 accum).
// ---------------------------------------------------------------------------

typedef __bf16 bf16_t;
typedef __attribute__((ext_vector_type(8)))  __bf16 v8bf;
typedef __attribute__((ext_vector_type(16))) __bf16 v16bf;
typedef __attribute__((ext_vector_type(8)))  float  v8f;

#define NB   16      // batches
#define NR   1024    // NS == NC
#define DD   128
#define DL   129     // D+1
#define PK   160     // 129 padded to multiple of 32 (bf16 rows)
#define KK   128
#define SQRT_KCONV 11.3137084989847603904f   // sqrt(128)

// ---------------- WMMA helpers ----------------

__device__ __forceinline__ v8f wmma_bf16(v16bf a, v16bf b, v8f c) {
  // (neg_a, A, neg_b, B, c_mod, C, reuse_a, reuse_b)
  return __builtin_amdgcn_wmma_f32_16x16x32_bf16(false, a, false, b, (short)0, c, false, false);
}

// A-frag: lane m holds row m; elems 0..7 = A[m][k0+kh*8 .. +7],
//         elems 8..15 = A[m][k0+16+kh*8 .. +7]. p = row + k0 + kh*8.
__device__ __forceinline__ v16bf load_a_frag(const bf16_t* p) {
  v8bf lo = *reinterpret_cast<const v8bf*>(p);
  v8bf hi = *reinterpret_cast<const v8bf*>(p + 16);
  return __builtin_shufflevector(lo, hi, 0,1,2,3,4,5,6,7,8,9,10,11,12,13,14,15);
}

// B-frag: lane n holds column n; elems 0..7 = Bt[n][k0+kh*8 .. +7],
//         elems 8..15 = Bt[n][k0+16+kh*8 .. +7]. p = row + k0 + kh*8.
__device__ __forceinline__ v16bf load_b_frag(const bf16_t* p) {
  v8bf lo = *reinterpret_cast<const v8bf*>(p);
  v8bf hi = *reinterpret_cast<const v8bf*>(p + 16);
  return __builtin_shufflevector(lo, hi, 0,1,2,3,4,5,6,7,8,9,10,11,12,13,14,15);
}

// ---------------- block reduction helper ----------------

template <int NT>
__device__ __forceinline__ float block_reduce_sum(float v, float* sh) {
  int t = threadIdx.x;
  sh[t] = v;
  __syncthreads();
  #pragma unroll
  for (int s = NT / 2; s > 0; s >>= 1) {
    if (t < s) sh[t] += sh[t + s];
    __syncthreads();
  }
  float r = sh[0];
  __syncthreads();
  return r;
}

// ---------------- p2l: poincare -> lorentz ----------------
// X (rows,128) f32 -> L129 (rows,129) f32 [+ optional bf16 row padded to 160]
__global__ void p2l_kernel(const float* __restrict__ X,
                           float* __restrict__ L129,
                           bf16_t* __restrict__ Lbf160) {
  int row = blockIdx.x;
  int t = threadIdx.x;           // 128 threads
  __shared__ float red[128];
  float x = X[(long)row * DD + t];
  float x2 = block_reduce_sum<128>(x * x, red);
  float f = SQRT_KCONV / (1.f - x2 + 1e-6f);
  float v0 = f * (1.f + x2);
  float vs = f * 2.f * x;
  float* o = L129 + (long)row * DL;
  if (t == 0) o[0] = v0;
  o[t + 1] = vs;
  if (Lbf160) {
    bf16_t* ob = Lbf160 + (long)row * PK;
    if (t == 0) ob[0] = (bf16_t)v0;
    ob[t + 1] = (bf16_t)vs;
    if (t < PK - DL) ob[DL + t] = (bf16_t)0.f;   // zero pad 129..159
  }
}

// ---------------- lorentz_linear (129 -> 129) ----------------
// mode 0: write bf16 padded row (row,160)           (Lw)
// mode 1: l2p -> outP f32 (row,128) + outPT bf16 transposed (128,1024) per batch
__global__ void lorentz_linear_kernel(const float* __restrict__ X,   // (rows,129)
                                      const float* __restrict__ W,   // (129,129)
                                      const float* __restrict__ bias,// (129)
                                      const float* __restrict__ scale_p,
                                      int mode,
                                      bf16_t* __restrict__ out_bf,
                                      float* __restrict__ outP,
                                      bf16_t* __restrict__ outPT) {
  int row = blockIdx.x;
  int t = threadIdx.x;            // 256 threads
  __shared__ float xs[DL];
  __shared__ float red[256];
  __shared__ float time_sh;
  const float* x = X + (long)row * DL;
  if (t < DL) xs[t] = x[t];
  __syncthreads();
  float y = 0.f;
  if (t < DL) {
    const float* w = W + (long)t * DL;
    float acc = bias[t];
    #pragma unroll 4
    for (int d = 0; d < DL; ++d) acc = fmaf(xs[d], w[d], acc);
    y = acc;
  }
  if (t == 0) {
    time_sh = 1.f / (1.f + expf(-y)) * expf(scale_p[0]) + 1.1f;
  }
  float contrib = (t >= 1 && t < DL) ? y * y : 0.f;
  float s2 = block_reduce_sum<256>(contrib, red);
  s2 = fmaxf(s2, 1e-8f);
  float tm = time_sh;
  float fac = sqrtf((tm * tm - 1.f) / s2);
  if (mode == 0) {
    bf16_t* o = out_bf + (long)row * PK;
    if (t < PK) {
      float v = (t == 0) ? tm : ((t < DL) ? y * fac : 0.f);
      o[t] = (bf16_t)v;
    }
  } else {
    if (t >= 1 && t < DL) {
      float v = (y * fac) / (tm + SQRT_KCONV);
      int j = t - 1;
      int b = row >> 10, n = row & (NR - 1);
      outP[(long)row * KK + j] = v;
      outPT[((long)b * KK + j) * NR + n] = (bf16_t)v;
    }
  }
}

// ---------------- GEMM1: L_raw = ls(1024x160) * Lw^T, tanh epilogue ----------
// Writes tanh(raw) bf16 to both L (s,c) and LT (c,s). Column 0 fixed later.
__global__ __launch_bounds__(256)
void wmma_gemm_tanh_kernel(const bf16_t* __restrict__ A,   // (b,1024,160)
                           const bf16_t* __restrict__ Bt,  // (b,1024,160)
                           bf16_t* __restrict__ L,         // (b,1024,1024)
                           bf16_t* __restrict__ LT,        // (b,1024,1024)
                           int total_waves) {
  int gw = (blockIdx.x * blockDim.x + threadIdx.x) >> 5;
  if (gw >= total_waves) return;
  int lane = threadIdx.x & 31;
  const int Nchunks = NR / 64;          // 16
  const int per_batch = (NR / 16) * Nchunks;
  int b  = gw / per_batch;
  int rem = gw - b * per_batch;
  int mt = rem / Nchunks;
  int nc = rem - mt * Nchunks;
  int lm = lane & 15, kh = lane >> 4;

  const bf16_t* Ab = A  + (long)b * NR * PK;
  const bf16_t* Bb = Bt + (long)b * NR * PK;
  const bf16_t* arow = Ab + (long)(mt * 16 + lm) * PK + kh * 8;
  const bf16_t* br[4];
  #pragma unroll
  for (int j = 0; j < 4; ++j)
    br[j] = Bb + (long)(nc * 64 + j * 16 + lm) * PK + kh * 8;

  v8f acc[4] = {};
  #pragma unroll
  for (int k0 = 0; k0 < PK; k0 += 32) {
    v16bf a = load_a_frag(arow + k0);
    #pragma unroll
    for (int j = 0; j < 4; ++j)
      acc[j] = wmma_bf16(a, load_b_frag(br[j] + k0), acc[j]);
  }

  long off = (long)b * NR * NR;
  int gm0 = mt * 16 + 8 * kh;
  #pragma unroll
  for (int j = 0; j < 4; ++j) {
    int gn = nc * 64 + j * 16 + lm;
    #pragma unroll
    for (int r = 0; r < 8; ++r) {
      float v = tanhf(acc[j][r]);
      int gm = gm0 + r;
      L [off + (long)gm * NR + gn] = (bf16_t)v;
      LT[off + (long)gn * NR + gm] = (bf16_t)v;
    }
  }
}

// ---------------- GEMM2/3: C(1024x128) = A(1024x1024) * Bt(128x1024)^T ------
__global__ __launch_bounds__(256)
void wmma_gemm_1024x128_kernel(const bf16_t* __restrict__ A,   // (b,1024,1024)
                               const bf16_t* __restrict__ Bt,  // (b,128,1024)
                               float* __restrict__ C,          // (b,1024,128)
                               int total_waves) {
  int gw = (blockIdx.x * blockDim.x + threadIdx.x) >> 5;
  if (gw >= total_waves) return;
  int lane = threadIdx.x & 31;
  const int Nchunks = KK / 64;          // 2
  const int per_batch = (NR / 16) * Nchunks;
  int b  = gw / per_batch;
  int rem = gw - b * per_batch;
  int mt = rem / Nchunks;
  int nc = rem - mt * Nchunks;
  int lm = lane & 15, kh = lane >> 4;

  const bf16_t* Ab = A  + (long)b * NR * NR;
  const bf16_t* Bb = Bt + (long)b * KK * NR;
  const bf16_t* arow = Ab + (long)(mt * 16 + lm) * NR + kh * 8;
  const bf16_t* br[4];
  #pragma unroll
  for (int j = 0; j < 4; ++j)
    br[j] = Bb + (long)(nc * 64 + j * 16 + lm) * NR + kh * 8;

  v8f acc[4] = {};
  for (int k0 = 0; k0 < NR; k0 += 32) {
    v16bf a = load_a_frag(arow + k0);
    #pragma unroll
    for (int j = 0; j < 4; ++j)
      acc[j] = wmma_bf16(a, load_b_frag(br[j] + k0), acc[j]);
  }

  float* Cb = C + (long)b * NR * KK;
  int gm0 = mt * 16 + 8 * kh;
  #pragma unroll
  for (int j = 0; j < 4; ++j) {
    int gn = nc * 64 + j * 16 + lm;
    #pragma unroll
    for (int r = 0; r < 8; ++r)
      Cb[(long)(gm0 + r) * KK + gn] = acc[j][r];
  }
}

// ---------------- time-component fixup + Ln ----------------
// Per (b,s) row of L: sum = sum_{c>=1} L^2 ; time = sqrt(1+sum)
// L[s][0] = LT[0][s] = time ; Ln[s] = sqrt(1 + 2*sum)
__global__ void lfix_kernel(bf16_t* __restrict__ L, bf16_t* __restrict__ LT,
                            float* __restrict__ Ln) {
  int row = blockIdx.x;               // b*1024 + s
  int t = threadIdx.x;                // 256
  __shared__ float red[256];
  const bf16_t* lr = L + (long)row * NR;
  float s = 0.f;
  for (int c = t; c < NR; c += 256)
    if (c >= 1) { float v = (float)lr[c]; s += v * v; }
  float sum = block_reduce_sum<256>(s, red);
  if (t == 0) {
    float time = sqrtf(1.f + sum);
    int b = row >> 10, sidx = row & (NR - 1);
    L[(long)row * NR] = (bf16_t)time;
    LT[(long)b * NR * NR + sidx] = (bf16_t)time;     // row c=0 of LT
    Ln[row] = sqrtf(fmaxf(1.f + 2.f * sum, 1e-30f));
  }
}

// ---------------- column norms: LTn[b][c] = ||L[:, c]|| ----------------
__global__ void ltn_kernel(const bf16_t* __restrict__ LT, float* __restrict__ LTn) {
  int row = blockIdx.x;               // b*1024 + c
  int t = threadIdx.x;                // 256
  __shared__ float red[256];
  const bf16_t* lr = LT + (long)row * NR;
  float s = 0.f;
  for (int i = t; i < NR; i += 256) { float v = (float)lr[i]; s += v * v; }
  float sum = block_reduce_sum<256>(s, red);
  if (t == 0) LTn[row] = sqrtf(fmaxf(sum, 1e-30f));
}

// ---------------- fused mobius/p2l/lorentz_tanh/logit ----------------
// One block per row. P,M: (rows,128) f32. Xn: (rows). wh: (129). logits: (rows)
__global__ void hsc_kernel(const float* __restrict__ P,
                           const float* __restrict__ M,
                           const float* __restrict__ Xn,
                           const float* __restrict__ wh,
                           float* __restrict__ logits) {
  int row = blockIdx.x;
  int t = threadIdx.x;                // 128
  __shared__ float red[128];
  float p = P[(long)row * KK + t];
  float m = M[(long)row * KK + t];

  // mobius_scale(M, xn)
  float mn2 = block_reduce_sum<128>(m * m, red);
  float mn = fmaxf(sqrtf(mn2), 1e-15f);
  float xn = Xn[row];
  float xnc = fminf(fmaxf(xn, -1.f + 1e-5f), 1.f - 1e-5f);
  float g = tanhf(mn / xn * atanhf(xnc)) / mn;
  float sc = g * m;

  // mobius_add(P, sc)
  float xy = block_reduce_sum<128>(p * sc, red);
  float x2 = block_reduce_sum<128>(p * p, red);
  float y2 = block_reduce_sum<128>(sc * sc, red);
  float num = (1.f + 2.f * xy + y2) * p + (1.f - x2) * sc;
  float den = fmaxf(1.f + 2.f * xy + x2 * y2, 1e-15f);
  float y = num / den;

  // p2l then lorentz_tanh (time component of p2l is discarded by lorentz_tanh)
  float yy = block_reduce_sum<128>(y * y, red);
  float f = SQRT_KCONV / (1.f - yy + 1e-6f);
  float h = tanhf(f * 2.f * y);                 // space components of H
  float hs2 = block_reduce_sum<128>(h * h, red);
  float time = sqrtf(1.f + hs2);

  // logit = mobius_scale(H . wh, ||H||)   (1-D vector case)
  float dotv = block_reduce_sum<128>(h * wh[t + 1], red) + time * wh[0];
  if (t == 0) {
    float Hn = sqrtf(fmaxf(2.f * time * time - 1.f, 1e-30f));   // ||H||
    float an = fmaxf(fabsf(dotv), 1e-15f);
    float artH = atanhf(fminf(fmaxf(Hn, -1.f + 1e-5f), 1.f - 1e-5f));
    logits[row] = tanhf(an / Hn * artH) * dotv / an;
  }
}

// ---------------- softmax over 1024, per batch ----------------
__global__ void softmax_kernel(const float* __restrict__ logits, float* __restrict__ out) {
  int b = blockIdx.x;
  int t = threadIdx.x;                // 256
  __shared__ float red[256];
  const float* lr = logits + (long)b * NR;
  float mx = -1e30f;
  for (int i = t; i < NR; i += 256) mx = fmaxf(mx, lr[i]);
  red[t] = mx; __syncthreads();
  #pragma unroll
  for (int s = 128; s > 0; s >>= 1) {
    if (t < s) red[t] = fmaxf(red[t], red[t + s]);
    __syncthreads();
  }
  mx = red[0]; __syncthreads();
  float se = 0.f;
  for (int i = t; i < NR; i += 256) se += expf(lr[i] - mx);
  float sum = block_reduce_sum<256>(se, red);
  float inv = 1.f / sum;
  for (int i = t; i < NR; i += 256) out[(long)b * NR + i] = expf(lr[i] - mx) * inv;
}

// ---------------- lorentz centroid ----------------
// co[b][d] = (sum_n A[n]*X[n][d]) / sqrt(clip(|inner|,1e-8))
__global__ void centroid_kernel(const float* __restrict__ Aw,   // (b,1024)
                                const float* __restrict__ X,    // (b,1024,129)
                                float* __restrict__ co) {       // (b,129)
  int b = blockIdx.x;
  int t = threadIdx.x;                // 256
  __shared__ float red[256];
  float avg = 0.f;
  if (t < DL) {
    const float* a = Aw + (long)b * NR;
    const float* x = X + (long)b * NR * DL + t;
    for (int n = 0; n < NR; ++n) avg = fmaf(a[n], x[(long)n * DL], avg);
  }
  float contrib = (t >= 1 && t < DL) ? avg * avg : ((t == 0) ? -avg * avg : 0.f);
  float inner = block_reduce_sum<256>(contrib, red);
  float den = sqrtf(fmaxf(fabsf(inner), 1e-8f));
  if (t < DL) co[(long)b * DL + t] = avg / den;
}

// ---------------- final assembly: co_sc (16,257) ----------------
__global__ void final_kernel(const float* __restrict__ co_s,
                             const float* __restrict__ co_c,
                             float* __restrict__ out) {
  int b = blockIdx.x;
  int t = threadIdx.x;                // 256
  __shared__ float red[256];
  float sp = (t < 128) ? co_s[(long)b * DL + 1 + t]
                       : co_c[(long)b * DL + 1 + (t - 128)];
  float s2 = block_reduce_sum<256>(sp * sp, red);
  if (t == 0) out[(long)b * 257] = sqrtf(1.f + s2);
  out[(long)b * 257 + 1 + t] = sp;
}

// ---------------------------------------------------------------------------

extern "C" void kernel_launch(void* const* d_in, const int* in_sizes, int n_in,
                              void* d_out, int out_size, void* d_ws, size_t ws_size,
                              hipStream_t stream) {
  (void)in_sizes; (void)n_in; (void)out_size; (void)ws_size;
  const float* sent  = (const float*)d_in[0];   // (16,1024,128)
  const float* comm  = (const float*)d_in[1];   // (16,1024,128)
  const float* Wl_w  = (const float*)d_in[2];
  const float* Wl_b  = (const float*)d_in[3];
  const float* Wl_s  = (const float*)d_in[4];
  const float* Ws_w  = (const float*)d_in[5];
  const float* Ws_b  = (const float*)d_in[6];
  const float* Ws_s  = (const float*)d_in[7];
  const float* Wc_w  = (const float*)d_in[8];
  const float* Wc_b  = (const float*)d_in[9];
  const float* Wc_s  = (const float*)d_in[10];
  const float* whs   = (const float*)d_in[11];
  const float* whc   = (const float*)d_in[12];
  float* out = (float*)d_out;

  // ---- workspace layout (all 256B aligned) ----
  char* w = (char*)d_ws;
  auto take = [&](size_t bytes) { char* p = w; w += (bytes + 255) & ~(size_t)255; return p; };
  float*  ls_f   = (float*) take((size_t)NB*NR*DL*4);
  float*  lc_f   = (float*) take((size_t)NB*NR*DL*4);
  bf16_t* ls_bf  = (bf16_t*)take((size_t)NB*NR*PK*2);
  bf16_t* Lw_bf  = (bf16_t*)take((size_t)NB*NR*PK*2);
  float*  Ps_f   = (float*) take((size_t)NB*NR*KK*4);
  float*  Pc_f   = (float*) take((size_t)NB*NR*KK*4);
  bf16_t* PsT    = (bf16_t*)take((size_t)NB*KK*NR*2);
  bf16_t* PcT    = (bf16_t*)take((size_t)NB*KK*NR*2);
  bf16_t* Lbf    = (bf16_t*)take((size_t)NB*NR*NR*2);
  bf16_t* LTbf   = (bf16_t*)take((size_t)NB*NR*NR*2);
  float*  Ln     = (float*) take((size_t)NB*NR*4);
  float*  LTn    = (float*) take((size_t)NB*NR*4);
  float*  Ms     = (float*) take((size_t)NB*NR*KK*4);
  float*  Mc     = (float*) take((size_t)NB*NR*KK*4);
  float*  log_s  = (float*) take((size_t)NB*NR*4);
  float*  log_c  = (float*) take((size_t)NB*NR*4);
  float*  co_s   = (float*) take((size_t)NB*DL*4);
  float*  co_c   = (float*) take((size_t)NB*DL*4);

  // d_out layout: co_sc (16*257) | As (16*1024) | Ac (16*1024)
  float* out_cosc = out;
  float* out_As   = out + NB * 257;
  float* out_Ac   = out_As + NB * NR;

  const int rows = NB * NR;            // 16384

  // 1. poincare -> lorentz
  p2l_kernel<<<rows, 128, 0, stream>>>(sent, ls_f, ls_bf);
  p2l_kernel<<<rows, 128, 0, stream>>>(comm, lc_f, nullptr);

  // 2. lorentz_linear layers
  lorentz_linear_kernel<<<rows, 256, 0, stream>>>(lc_f, Wl_w, Wl_b, Wl_s, 0, Lw_bf, nullptr, nullptr);
  lorentz_linear_kernel<<<rows, 256, 0, stream>>>(ls_f, Ws_w, Ws_b, Ws_s, 1, nullptr, Ps_f, PsT);
  lorentz_linear_kernel<<<rows, 256, 0, stream>>>(lc_f, Wc_w, Wc_b, Wc_s, 1, nullptr, Pc_f, PcT);

  // 3. GEMM1 (WMMA bf16) with tanh epilogue -> L, LT
  {
    int waves = NB * (NR / 16) * (NR / 64);          // 16384
    int blocks = waves * 32 / 256;
    wmma_gemm_tanh_kernel<<<blocks, 256, 0, stream>>>(ls_bf, Lw_bf, Lbf, LTbf, waves);
  }

  // 4. time-component fixup + norms
  lfix_kernel<<<rows, 256, 0, stream>>>(Lbf, LTbf, Ln);
  ltn_kernel <<<rows, 256, 0, stream>>>(LTbf, LTn);

  // 5. GEMM2 / GEMM3 (WMMA bf16)
  {
    int waves = NB * (NR / 16) * (KK / 64);          // 2048
    int blocks = waves * 32 / 256;
    wmma_gemm_1024x128_kernel<<<blocks, 256, 0, stream>>>(Lbf,  PcT, Ms, waves);
    wmma_gemm_1024x128_kernel<<<blocks, 256, 0, stream>>>(LTbf, PsT, Mc, waves);
  }

  // 6. fused mobius/p2l/tanh/logit
  hsc_kernel<<<rows, 128, 0, stream>>>(Ps_f, Ms, Ln,  whs, log_s);
  hsc_kernel<<<rows, 128, 0, stream>>>(Pc_f, Mc, LTn, whc, log_c);

  // 7. softmax -> As, Ac (directly into d_out)
  softmax_kernel<<<NB, 256, 0, stream>>>(log_s, out_As);
  softmax_kernel<<<NB, 256, 0, stream>>>(log_c, out_Ac);

  // 8. centroids + final output
  centroid_kernel<<<NB, 256, 0, stream>>>(out_As, ls_f, co_s);
  centroid_kernel<<<NB, 256, 0, stream>>>(out_Ac, lc_f, co_c);
  final_kernel<<<NB, 256, 0, stream>>>(co_s, co_c, out_cosc);
}
